// FourierBlock_85907935855002
// MI455X (gfx1250) — matrix operand order repaired and checked by reading, not compiled, not run on hardware
//
#include <hip/hip_runtime.h>

typedef _Float16 v16h __attribute__((ext_vector_type(16)));
typedef _Float16 v8h  __attribute__((ext_vector_type(8)));
typedef float    v8f  __attribute__((ext_vector_type(8)));

#define Bdim 8
#define Ldim 4096
#define Hdim 8
#define Edim 64
#define Mdim 64
#define BH   64   // Bdim*Hdim

// workspace layout (bytes)
#define OFF_TFC (0u)            // half[64][4096]  cos
#define OFF_TFS (512u<<10)      // half[64][4096]  -sin
#define OFF_TIC (1024u<<10)     // half[4096][64]  (m?2:1)/L * cos
#define OFF_TIS (1536u<<10)     // half[4096][64]  -(2/L) * sin
#define OFF_XR  (2048u<<10)     // float[64][64][64] (bh,e,m)
#define OFF_XI  (3072u<<10)
#define OFF_YR  (4096u<<10)     // half[64][64][64] (bh,o,m)
#define OFF_YI  (4608u<<10)
// total 5 MB

// ---------------------------------------------------------------------------
// Kernel 0: DFT basis tables (f16). Phase uses exact integer mod -> no
// argument-reduction error: theta = 2*pi*((m*n) mod L)/L.
// ---------------------------------------------------------------------------
__global__ void __launch_bounds__(256) k_tables(_Float16* __restrict__ tfc,
                                                _Float16* __restrict__ tfs,
                                                _Float16* __restrict__ tic,
                                                _Float16* __restrict__ tis) {
  unsigned idx = blockIdx.x * 256u + threadIdx.x;   // 0 .. 64*4096-1
  unsigned n = idx & (Ldim - 1u);
  unsigned m = idx >> 12;                           // 0..63
  float ph = (float)((m * n) & (Ldim - 1u)) * (6.28318530717958647692f / (float)Ldim);
  float s, c;
  __sincosf(ph, &s, &c);
  tfc[m * Ldim + n] = (_Float16)c;
  tfs[m * Ldim + n] = (_Float16)(-s);
  float scc = (m ? 2.0f : 1.0f) * (1.0f / (float)Ldim);
  tic[n * Mdim + m] = (_Float16)(c * scc);
  tis[n * Mdim + m] = (_Float16)(-s * (2.0f / (float)Ldim));
}

// ---------------------------------------------------------------------------
// Kernel 1: forward truncated DFT via WMMA f16 -> f32.
// Per (b,h): X[e,m] = sum_n q[b,n,h,e] * basis[n,m].
// grid = 128 blocks = (bh, part{re,im}); 8 waves: e_tile = w&3, m_pair = w>>2.
// A = q tile (M=e rows, K=n), B = basis table row-major [m][n] (N=m, K=n).
// ---------------------------------------------------------------------------
__global__ void __launch_bounds__(256) k_forward(const float* __restrict__ q,
                                                 const _Float16* __restrict__ tfc,
                                                 const _Float16* __restrict__ tfs,
                                                 float* __restrict__ xr,
                                                 float* __restrict__ xi) {
  const unsigned tid = threadIdx.x;
  const unsigned lane = tid & 31u, wv = tid >> 5;
  const unsigned part = blockIdx.x & 1u, bh = blockIdx.x >> 1;
  const unsigned b = bh >> 3, h = bh & 7u;
  const unsigned e_tile = wv & 3u, m_pair = wv >> 2;   // m_pair in 0..1
  const unsigned lane16 = lane & 15u, hi = lane >> 4;  // lane half
  const unsigned e_row = e_tile * 16u + lane16;        // A: M = lane%16
  const unsigned koff = hi * 8u;                       // A: K group offset (0 or 8)
  const unsigned nofs = hi * 8u;                       // B: K group offset (0 or 8)
  const _Float16* tab = part ? tfs : tfc;

  // q element (b,n,h,e) at ((b*L + n)*H + h)*E + e ; stride over n = H*E = 512
  const float* qbase = q + (size_t)b * Ldim * Hdim * Edim + (size_t)h * Edim + e_row;

  v8f acc0 = {};
  v8f acc1 = {};
  const unsigned mcol_a = (m_pair * 2u + 0u) * 16u + lane16;  // B: N = lane%16
  const unsigned mcol_b = (m_pair * 2u + 1u) * 16u + lane16;

  for (unsigned n0 = 0; n0 < Ldim; n0 += 32u) {
    // ---- A fragment: 16 strided q loads (coalesced across lanes over e), cvt f32->f16
    const float* qq = qbase + (size_t)(n0 + koff) * (Hdim * Edim);
    v16h A;
#pragma unroll
    for (int v = 0; v < 4; ++v) {              // VGPR v: K = koff+2v, koff+2v+1
      A[2 * v]     = (_Float16)qq[(2 * v) * (Hdim * Edim)];
      A[2 * v + 1] = (_Float16)qq[(2 * v + 1) * (Hdim * Edim)];
    }
    const float* qq2 = qq + 16 * (Hdim * Edim); // VGPR 4..7: K = 16+koff+...
#pragma unroll
    for (int v = 0; v < 4; ++v) {
      A[8 + 2 * v]     = (_Float16)qq2[(2 * v) * (Hdim * Edim)];
      A[8 + 2 * v + 1] = (_Float16)qq2[(2 * v + 1) * (Hdim * Edim)];
    }
    if (n0 + 32u < Ldim)
      __builtin_prefetch(qbase + (size_t)(n0 + 32u + koff) * (Hdim * Edim), 0, 1);

    // ---- B fragments from basis table: two runs of 8 contiguous halves per lane, 16 apart
    {
      const v8h* bp = (const v8h*)(tab + (size_t)mcol_a * Ldim + n0 + nofs);
      v8h b0 = bp[0], b1 = bp[2];
      v16h Bv;
#pragma unroll
      for (int i = 0; i < 8; ++i) { Bv[i] = b0[i]; Bv[8 + i] = b1[i]; }
      acc0 = __builtin_amdgcn_wmma_f32_16x16x32_f16(false, A, false, Bv, (short)0,
                                                    acc0, false, false);
    }
    {
      const v8h* bp = (const v8h*)(tab + (size_t)mcol_b * Ldim + n0 + nofs);
      v8h b0 = bp[0], b1 = bp[2];
      v16h Bv;
#pragma unroll
      for (int i = 0; i < 8; ++i) { Bv[i] = b0[i]; Bv[8 + i] = b1[i]; }
      acc1 = __builtin_amdgcn_wmma_f32_16x16x32_f16(false, A, false, Bv, (short)0,
                                                    acc1, false, false);
    }
  }

  // ---- store D: row M = e_tile*16 + r + 8*hi, col N already in mcol_*
  float* X = part ? xi : xr;
#pragma unroll
  for (int r = 0; r < 8; ++r) {
    unsigned e = e_tile * 16u + (unsigned)r + hi * 8u;
    X[((size_t)bh * 64u + e) * 64u + mcol_a] = acc0[r];
    X[((size_t)bh * 64u + e) * 64u + mcol_b] = acc1[r];
  }
}

// ---------------------------------------------------------------------------
// Kernel 2: complex mode mixing in f32 (W entries ~4e-6 stay exact in f32).
// Y[bh,o,m] = sum_e X[bh,e,m] * W[m,h,e,o]  (complex). Lanes over o so W reads
// are coalesced; X values are wave-uniform broadcasts. Output stored f16.
// ---------------------------------------------------------------------------
__global__ void __launch_bounds__(256) k_mix(const float* __restrict__ xr,
                                             const float* __restrict__ xi,
                                             const float* __restrict__ wr,
                                             const float* __restrict__ wi,
                                             _Float16* __restrict__ yr,
                                             _Float16* __restrict__ yi) {
  unsigned bh = blockIdx.x >> 4;
  unsigned m = (blockIdx.x & 15u) * 4u + (threadIdx.x >> 6);
  unsigned o = threadIdx.x & 63u;
  unsigned h = bh & 7u;
  const float* xrp = xr + (size_t)bh * 64u * 64u + m;
  const float* xip = xi + (size_t)bh * 64u * 64u + m;
  const float* wrp = wr + (((size_t)m * Hdim + h) * Edim) * Edim + o;
  const float* wip = wi + (((size_t)m * Hdim + h) * Edim) * Edim + o;
  float ar = 0.0f, ai = 0.0f;
#pragma unroll 4
  for (int e = 0; e < Edim; ++e) {
    float xrv = xrp[e * 64], xiv = xip[e * 64];
    float wrv = wrp[e * 64], wiv = wip[e * 64];
    ar += xrv * wrv - xiv * wiv;
    ai += xrv * wiv + xiv * wrv;
  }
  yr[((size_t)bh * 64u + o) * 64u + m] = (_Float16)ar;
  yi[((size_t)bh * 64u + o) * 64u + m] = (_Float16)ai;
}

// ---------------------------------------------------------------------------
// Kernel 3: truncated irfft via WMMA.
// out[n,o] = sum_m ( Cinv[n,m]*Yr[o,m] + Sinv[n,m]*Yi[o,m] ), scales baked
// into tables (DC weight 1/L, others 2/L, -sin). grid = (bh, 16 chunks of 256n),
// 8 waves: n_tile = w&3, o_half = w>>2; 4 reps of 64 n each; B kept in regs.
// ---------------------------------------------------------------------------
__global__ void __launch_bounds__(256) k_inverse(const _Float16* __restrict__ tic,
                                                 const _Float16* __restrict__ tis,
                                                 const _Float16* __restrict__ yr,
                                                 const _Float16* __restrict__ yi,
                                                 float* __restrict__ out) {
  unsigned tid = threadIdx.x, lane = tid & 31u, wv = tid >> 5;
  unsigned bh = blockIdx.x >> 4, chunk = blockIdx.x & 15u;
  unsigned b = bh >> 3, h = bh & 7u;
  unsigned o_half = wv >> 2, n_t = wv & 3u;
  unsigned lane16 = lane & 15u, hi = lane >> 4;
  unsigned moffB = hi * 8u;    // B K-group offset
  unsigned koffA = hi * 8u;    // A K-group offset

  // ---- preload B fragments (Y) : [K=m, N=o], m contiguous per o row
  v16h Br[2][2], Bi[2][2];
#pragma unroll
  for (int s = 0; s < 2; ++s) {
#pragma unroll
    for (int ot = 0; ot < 2; ++ot) {
      unsigned o = o_half * 32u + (unsigned)ot * 16u + lane16;
      const v8h* pr = (const v8h*)(yr + ((size_t)bh * 64u + o) * 64u + s * 32u + moffB);
      const v8h* pi = (const v8h*)(yi + ((size_t)bh * 64u + o) * 64u + s * 32u + moffB);
      v8h r0 = pr[0], r1 = pr[2], i0 = pi[0], i1 = pi[2];
#pragma unroll
      for (int j = 0; j < 8; ++j) {
        Br[s][ot][j] = r0[j]; Br[s][ot][8 + j] = r1[j];
        Bi[s][ot][j] = i0[j]; Bi[s][ot][8 + j] = i1[j];
      }
    }
  }

  for (unsigned rep = 0; rep < 4u; ++rep) {
    unsigned nbase = chunk * 256u + rep * 64u + n_t * 16u;
    unsigned nrow = nbase + lane16;    // A: M = lane%16
    v8f a0 = {}, a1 = {};
#pragma unroll
    for (int s = 0; s < 2; ++s) {
      // A fragments: halves K = s*32+koffA+0..7 and s*32+16+koffA+0..7
      const v8h* pc = (const v8h*)(tic + (size_t)nrow * 64u + s * 32u + koffA);
      const v8h* ps = (const v8h*)(tis + (size_t)nrow * 64u + s * 32u + koffA);
      v8h c0 = pc[0], c1 = pc[2], s0 = ps[0], s1 = ps[2];  // +32B = +16 halves
      v16h Ac, As;
#pragma unroll
      for (int j = 0; j < 8; ++j) {
        Ac[j] = c0[j]; Ac[8 + j] = c1[j];
        As[j] = s0[j]; As[8 + j] = s1[j];
      }
      a0 = __builtin_amdgcn_wmma_f32_16x16x32_f16(false, Ac, false, Br[s][0], (short)0, a0, false, false);
      a0 = __builtin_amdgcn_wmma_f32_16x16x32_f16(false, As, false, Bi[s][0], (short)0, a0, false, false);
      a1 = __builtin_amdgcn_wmma_f32_16x16x32_f16(false, Ac, false, Br[s][1], (short)0, a1, false, false);
      a1 = __builtin_amdgcn_wmma_f32_16x16x32_f16(false, As, false, Bi[s][1], (short)0, a1, false, false);
    }
    // ---- store to out[b,n,h,o] (o contiguous across lanes 0-15)
#pragma unroll
    for (int r = 0; r < 8; ++r) {
      unsigned n = nbase + (unsigned)r + hi * 8u;
      size_t base = ((size_t)(b * Ldim + n) * Hdim + h) * Edim;
      out[base + o_half * 32u + lane16]       = a0[r];
      out[base + o_half * 32u + 16u + lane16] = a1[r];
    }
  }
}

// ---------------------------------------------------------------------------
extern "C" void kernel_launch(void* const* d_in, const int* in_sizes, int n_in,
                              void* d_out, int out_size, void* d_ws, size_t ws_size,
                              hipStream_t stream) {
  (void)in_sizes; (void)n_in; (void)out_size; (void)ws_size;
  const float* q  = (const float*)d_in[0];
  // d_in[1]=k, d_in[2]=v, d_in[3]=mask are unused by the reference math
  const float* wr = (const float*)d_in[4];
  const float* wi = (const float*)d_in[5];
  float* out = (float*)d_out;
  char* ws = (char*)d_ws;

  _Float16* tfc = (_Float16*)(ws + OFF_TFC);
  _Float16* tfs = (_Float16*)(ws + OFF_TFS);
  _Float16* tic = (_Float16*)(ws + OFF_TIC);
  _Float16* tis = (_Float16*)(ws + OFF_TIS);
  float*    xr  = (float*)(ws + OFF_XR);
  float*    xi  = (float*)(ws + OFF_XI);
  _Float16* yrp = (_Float16*)(ws + OFF_YR);
  _Float16* yip = (_Float16*)(ws + OFF_YI);

  k_tables <<<1024, 256, 0, stream>>>(tfc, tfs, tic, tis);
  k_forward<<< 128, 256, 0, stream>>>(q, tfc, tfs, xr, xi);
  k_mix    <<<1024, 256, 0, stream>>>(xr, xi, wr, wi, yrp, yip);
  k_inverse<<<1024, 256, 0, stream>>>(tic, tis, yrp, yip, out);
}
